// Classifier7_54022098649414
// MI455X (gfx1250) — matrix operand order repaired and checked by reading, not compiled, not run on hardware
//
#include <hip/hip_runtime.h>
#include <hip/hip_bf16.h>
#include <math.h>

// ---------------------------------------------------------------------------
// GNN classifier for MI455X (gfx1250, wave32).
// Dense GEMMs (dilated-conv-as-GEMM, GCN/JK feature transforms) run on the
// WMMA pipe via __builtin_amdgcn_wmma_f32_16x16x32_f16 with pre-packed f16
// B fragments. Edge aggregation is atomic scatter; BN stats are block-reduced.
// ---------------------------------------------------------------------------

typedef __attribute__((ext_vector_type(16))) _Float16 v16h;
typedef __attribute__((ext_vector_type(8)))  float    v8f;

#define N_NODES 50000
#define N_EDGES 800000
#define N_GRAPHS 64
#define EPS_F 1e-5f

__device__ __forceinline__ v8f v8f_zero() {
  v8f z;
#pragma unroll
  for (int i = 0; i < 8; ++i) z[i] = 0.f;
  return z;
}

// ---------------- utility kernels ----------------

__global__ void zero_kernel(float* __restrict__ p, int n) {
  for (int i = blockIdx.x * blockDim.x + threadIdx.x; i < n;
       i += gridDim.x * blockDim.x)
    p[i] = 0.f;
}

__global__ void f32_to_f16_kernel(const float* __restrict__ x,
                                  _Float16* __restrict__ y, int n) {
  for (int i = blockIdx.x * blockDim.x + threadIdx.x; i < n;
       i += gridDim.x * blockDim.x)
    y[i] = (_Float16)x[i];
}

// Pack a row-major [Kact x Nact] weight (row width = Nact) into WMMA B
// fragments: frag f = kt*NT + nt, element (lane,hh) -> K = kt*32 +
// (hh>>3)*16 + (lane>>4)*8 + (hh&7), n = nt*16 + (lane&15). Zero-pads K/N.
__global__ void pack_rowmajor_kernel(const float* __restrict__ W,
                                     _Float16* __restrict__ dst, int Kact,
                                     int Nact, int KT, int NT) {
  int total = KT * NT * 512;
  for (int idx = blockIdx.x * blockDim.x + threadIdx.x; idx < total;
       idx += gridDim.x * blockDim.x) {
    int f = idx >> 9, w = idx & 511;
    int lane = w >> 4, hh = w & 15;
    int kt = f / NT, nt = f - kt * NT;
    int k = kt * 32 + ((hh & 8) << 1) + ((lane >> 4) << 3) + (hh & 7);
    int n = nt * 16 + (lane & 15);
    float v = (k < Kact && n < Nact) ? W[k * Nact + n] : 0.f;
    dst[idx] = (_Float16)v;
  }
}

// conv_w [3][96][128][3]; B[K=ks*128+ci][n=co] per branch; 12 ktiles, 6 ntiles.
__global__ void pack_conv_kernel(const float* __restrict__ conv_w,
                                 _Float16* __restrict__ dst) {
  int total = 3 * 12 * 6 * 512;
  for (int idx = blockIdx.x * blockDim.x + threadIdx.x; idx < total;
       idx += gridDim.x * blockDim.x) {
    int f = idx >> 9, w = idx & 511;
    int lane = w >> 4, hh = w & 15;
    int nt = f % 6;
    int kt = (f / 6) % 12;
    int br = f / 72;
    int K = kt * 32 + ((hh & 8) << 1) + ((lane >> 4) << 3) + (hh & 7);  // 0..383
    int n = nt * 16 + (lane & 15);              // 0..95
    int ci = K & 127, ks = K >> 7;
    dst[idx] =
        (_Float16)conv_w[(((size_t)br * 96 + n) * 128 + ci) * 3 + ks];
  }
}

// ---------------- WMMA kernels ----------------

// Dilated conv as GEMM: one wave per (node-tile, branch). A gathered from
// the f16 node features at rows m-dil, m, m+dil (zero off-range).
__global__ void conv_wmma_kernel(const _Float16* __restrict__ x16,
                                 const _Float16* __restrict__ wpack,
                                 const float* __restrict__ conv_b,
                                 float* __restrict__ h0x, int nNodes) {
  int wave = (blockIdx.x * blockDim.x + threadIdx.x) >> 5;
  int lane = threadIdx.x & 31;
  int nRowTiles = nNodes >> 4;
  if (wave >= nRowTiles * 3) return;
  int br = wave % 3;
  int rt = wave / 3;
  int dil = br + 1;
  int m = rt * 16 + (lane & 15);
  int base8 = (lane >> 4) << 3;  // 0 or 8

  v8f acc[6];
#pragma unroll
  for (int j = 0; j < 6; ++j) acc[j] = v8f_zero();

  const _Float16* wb = wpack + (size_t)br * (12 * 6 * 512);
  for (int kk = 0; kk < 12; ++kk) {
    int s = kk >> 2;                 // conv tap 0..2
    int ci0 = (kk & 3) << 5;         // channel base within tap
    int node = m + (s - 1) * dil;
    v16h a;
    if ((unsigned)node < (unsigned)nNodes) {
      const _Float16* rp = x16 + (size_t)node * 128 + ci0 + base8;
#pragma unroll
      for (int j = 0; j < 8; ++j) {
        a[j] = rp[j];          // K_local = base8 + j
        a[j + 8] = rp[16 + j]; // K_local = 16 + base8 + j
      }
    } else {
#pragma unroll
      for (int j = 0; j < 16; ++j) a[j] = (_Float16)0.f;
    }
    const _Float16* fb = wb + (size_t)(kk * 6) * 512 + lane * 16;
#pragma unroll
    for (int nt = 0; nt < 6; ++nt) {
      v16h b = *(const v16h*)(fb + nt * 512);
      acc[nt] = __builtin_amdgcn_wmma_f32_16x16x32_f16(
          false, a, false, b, (short)0, acc[nt], false, false);
    }
  }

  float* outp = h0x + (size_t)br * nNodes * 96;
  int rowAdd = (lane >> 4) << 3;
  int col0 = lane & 15;
  const float* bb = conv_b + br * 96;
#pragma unroll
  for (int nt = 0; nt < 6; ++nt) {
    int n2 = nt * 16 + col0;
    float bias = bb[n2];
#pragma unroll
    for (int r = 0; r < 8; ++r) {
      int m2 = rt * 16 + r + rowAdd;
      outp[(size_t)m2 * 96 + n2] = acc[nt][r] + bias;
    }
  }
}

// Generic GEMM: C[N,Nact] = relu?(A[N,Kact] @ Bpack + bias). A is f32
// row-major (converted per-lane to f16), rows % 16 == 0. NT col-tiles.
template <int NT>
__global__ void wmma_gemm_kernel(const float* __restrict__ A, int lda,
                                 int Kact, int KT,
                                 const _Float16* __restrict__ Bpack, int Nact,
                                 const float* __restrict__ bias,
                                 float* __restrict__ C, int ldc,
                                 int nRowTiles, int doRelu) {
  int wave = (blockIdx.x * blockDim.x + threadIdx.x) >> 5;
  int lane = threadIdx.x & 31;
  if (wave >= nRowTiles) return;
  int m = wave * 16 + (lane & 15);
  int base8 = (lane >> 4) << 3;

  v8f acc[NT];
#pragma unroll
  for (int j = 0; j < NT; ++j) acc[j] = v8f_zero();

  const float* rp = A + (size_t)m * lda;
  for (int kt = 0; kt < KT; ++kt) {
    v16h a;
#pragma unroll
    for (int hh = 0; hh < 16; ++hh) {
      int kl = (hh < 8 ? hh : hh + 8) + base8;
      int k = kt * 32 + kl;
      a[hh] = (k < Kact) ? (_Float16)rp[k] : (_Float16)0.f;
    }
    const _Float16* fb = Bpack + (size_t)(kt * NT) * 512 + lane * 16;
#pragma unroll
    for (int nt = 0; nt < NT; ++nt) {
      v16h b = *(const v16h*)(fb + nt * 512);
      acc[nt] = __builtin_amdgcn_wmma_f32_16x16x32_f16(
          false, a, false, b, (short)0, acc[nt], false, false);
    }
  }

  int rowAdd = (lane >> 4) << 3;
  int col0 = lane & 15;
#pragma unroll
  for (int nt = 0; nt < NT; ++nt) {
    int n2 = nt * 16 + col0;
    if (n2 < Nact) {
      float bv = bias ? bias[n2] : 0.f;
#pragma unroll
      for (int r = 0; r < 8; ++r) {
        int m2 = wave * 16 + r + rowAdd;
        float v = acc[nt][r] + bv;
        if (doRelu) v = fmaxf(v, 0.f);
        C[(size_t)m2 * ldc + n2] = v;
      }
    }
  }
}

// ---------------- elementwise / reduction kernels ----------------

// t_i = relu(h0x[i]) + h0x[(i+1)%3], all [N,96]
__global__ void cross_t_kernel(const float* __restrict__ h0x,
                               float* __restrict__ t, int N) {
  int NF = N * 96;
  int total = 3 * NF;
  for (int idx = blockIdx.x * blockDim.x + threadIdx.x; idx < total;
       idx += gridDim.x * blockDim.x) {
    int b = idx / NF;
    int rem = idx - b * NF;
    int b2 = (b + 1) % 3;
    t[idx] = fmaxf(h0x[(size_t)b * NF + rem], 0.f) + h0x[(size_t)b2 * NF + rem];
  }
}

// per-channel sum & sumsq over rows; launch block(F,8), grid(ceil(N/512))
__global__ void bn_stats_kernel(const float* __restrict__ X,
                                float* __restrict__ stats, int N, int F) {
  __shared__ float sh1[8][96];
  __shared__ float sh2[8][96];
  int ch = threadIdx.x;
  int ty = threadIdx.y;
  int r0 = blockIdx.x * 512;
  int rend = r0 + 512;
  if (rend > N) rend = N;
  float s = 0.f, s2 = 0.f;
  for (int r = r0 + ty; r < rend; r += 8) {
    float v = X[(size_t)r * F + ch];
    s += v;
    s2 += v * v;
  }
  sh1[ty][ch] = s;
  sh2[ty][ch] = s2;
  __syncthreads();
  if (ty == 0) {
#pragma unroll
    for (int tt = 1; tt < 8; ++tt) {
      s += sh1[tt][ch];
      s2 += sh2[tt][ch];
    }
    atomicAdd(&stats[ch * 2], s);
    atomicAdd(&stats[ch * 2 + 1], s2);
  }
}

__global__ void bn_apply_kernel(const float* __restrict__ X,
                                float* __restrict__ Y,
                                const float* __restrict__ stats,
                                const float* __restrict__ g,
                                const float* __restrict__ b, int total, int F,
                                float invN) {
  for (int idx = blockIdx.x * blockDim.x + threadIdx.x; idx < total;
       idx += gridDim.x * blockDim.x) {
    int ch = idx % F;
    float m = stats[ch * 2] * invN;
    float var = stats[ch * 2 + 1] * invN - m * m;
    Y[idx] = g[ch] * (X[idx] - m) * rsqrtf(var + EPS_F) + b[ch];
  }
}

// agg[dst] += h[src] * ew ; one thread per (edge, channel)
__global__ void edge_agg_kernel(const int* __restrict__ src,
                                const int* __restrict__ dst,
                                const float* __restrict__ ew,
                                const float* __restrict__ h,
                                float* __restrict__ agg, int E, int F) {
  int total = E * F;
  for (int idx = blockIdx.x * blockDim.x + threadIdx.x; idx < total;
       idx += gridDim.x * blockDim.x) {
    int e = idx / F;
    int c = idx - e * F;
    float v = h[(size_t)src[e] * F + c] * ew[e];
    atomicAdd(&agg[(size_t)dst[e] * F + c], v);
  }
}

// pressed[n,f] = pb + sum_{i,k} pw[i*3+k] * h2x[i][n][f+k-1]
__global__ void press_kernel(const float* __restrict__ h2x,
                             const float* __restrict__ pw,
                             const float* __restrict__ pb,
                             float* __restrict__ pressed, int N) {
  int total = N * 54;
  for (int idx = blockIdx.x * blockDim.x + threadIdx.x; idx < total;
       idx += gridDim.x * blockDim.x) {
    int n = idx / 54;
    int f = idx - 54 * n;
    float acc = pb[0];
#pragma unroll
    for (int i = 0; i < 3; ++i)
#pragma unroll
      for (int k = 0; k < 3; ++k) {
        int ff = f + k - 1;
        if (ff >= 0 && ff < 54)
          acc += pw[i * 3 + k] * h2x[((size_t)i * N + n) * 54 + ff];
      }
    pressed[idx] = acc;
  }
}

// hg[gid[n]][colOff + c] += x[n][c]
__global__ void readout_kernel(const int* __restrict__ gid,
                               const float* __restrict__ x,
                               float* __restrict__ hg, int N, int F,
                               int colOff) {
  int total = N * F;
  for (int idx = blockIdx.x * blockDim.x + threadIdx.x; idx < total;
       idx += gridDim.x * blockDim.x) {
    int n = idx / F;
    int c = idx - n * F;
    atomicAdd(&hg[(size_t)gid[n] * 75 + colOff + c], x[idx]);
  }
}

// BN(hg) @ W1 + b1 @ W2 + b2 -> logits [64,10]; single block.
__global__ void classifier_kernel(const float* __restrict__ hg,
                                  const float* __restrict__ stats,
                                  const float* __restrict__ g,
                                  const float* __restrict__ b,
                                  const float* __restrict__ w1,
                                  const float* __restrict__ b1,
                                  const float* __restrict__ w2,
                                  const float* __restrict__ b2,
                                  float* __restrict__ out) {
  __shared__ float xs[64 * 75];
  __shared__ float hs[64 * 120];
  int tid = threadIdx.x;
  for (int idx = tid; idx < 64 * 75; idx += blockDim.x) {
    int ch = idx % 75;
    float m = stats[ch * 2] * (1.f / 64.f);
    float var = stats[ch * 2 + 1] * (1.f / 64.f) - m * m;
    xs[idx] = g[ch] * (hg[idx] - m) * rsqrtf(var + EPS_F) + b[ch];
  }
  __syncthreads();
  for (int idx = tid; idx < 64 * 120; idx += blockDim.x) {
    int gph = idx / 120, o = idx - 120 * gph;
    float acc = b1[o];
    for (int k = 0; k < 75; ++k) acc += xs[gph * 75 + k] * w1[k * 120 + o];
    hs[idx] = acc;
  }
  __syncthreads();
  for (int idx = tid; idx < 640; idx += blockDim.x) {
    int gph = idx / 10, o = idx - 10 * gph;
    float acc = b2[o];
    for (int k = 0; k < 120; ++k) acc += hs[gph * 120 + k] * w2[k * 10 + o];
    out[idx] = acc;
  }
}

// ---------------- host orchestration ----------------

static inline int gsBlocks(size_t total) {
  size_t b = (total + 255) / 256;
  if (b > 262144) b = 262144;
  return (int)b;
}

extern "C" void kernel_launch(void* const* d_in, const int* in_sizes, int n_in,
                              void* d_out, int out_size, void* d_ws,
                              size_t ws_size, hipStream_t stream) {
  const float* node_feats = (const float*)d_in[0];
  const float* edge_w = (const float*)d_in[1];
  const float* conv_w = (const float*)d_in[2];
  const float* conv_b = (const float*)d_in[3];
  const float* gcn_w = (const float*)d_in[4];
  const float* gcn_b = (const float*)d_in[5];
  const float* bn1_g = (const float*)d_in[6];
  const float* bn1_b = (const float*)d_in[7];
  const float* bn2_g = (const float*)d_in[8];
  const float* bn2_b = (const float*)d_in[9];
  const float* bn3_g = (const float*)d_in[10];
  const float* bn3_b = (const float*)d_in[11];
  const float* press_w = (const float*)d_in[12];
  const float* press_b = (const float*)d_in[13];
  const float* jk_w1 = (const float*)d_in[14];
  const float* jk_b1 = (const float*)d_in[15];
  const float* jk_w2 = (const float*)d_in[16];
  const float* jk_b2 = (const float*)d_in[17];
  const float* jk_w3 = (const float*)d_in[18];
  const float* jk_b3 = (const float*)d_in[19];
  const float* cls1_w = (const float*)d_in[20];
  const float* cls1_b = (const float*)d_in[21];
  const float* cls2_w = (const float*)d_in[22];
  const float* cls2_b = (const float*)d_in[23];
  const int* edge_src = (const int*)d_in[24];
  const int* edge_dst = (const int*)d_in[25];
  const int* graph_ids = (const int*)d_in[26];

  const int N = N_NODES, E = N_EDGES;
  float* logits = (float*)d_out;        // [64,10]
  float* emd = (float*)d_out + 640;     // [N,54]

  // workspace carve-up (~155 MB), 256B aligned
  size_t off = 0;
  char* ws = (char*)d_ws;
  auto alloc = [&](size_t bytes) -> void* {
    void* p = ws + off;
    off = (off + bytes + 255) & ~(size_t)255;
    return p;
  };
  _Float16* x16 = (_Float16*)alloc((size_t)N * 128 * 2);
  _Float16* packConv = (_Float16*)alloc((size_t)3 * 12 * 6 * 512 * 2);
  _Float16* packGcn = (_Float16*)alloc((size_t)3 * 3 * 4 * 512 * 2);
  _Float16* packJk1 = (_Float16*)alloc((size_t)2 * 3 * 512 * 2);
  _Float16* packJk2 = (_Float16*)alloc((size_t)2 * 2 * 512 * 2);
  _Float16* packJk3 = (_Float16*)alloc((size_t)1 * 1 * 512 * 2);
  float* regionA = (float*)alloc((size_t)3 * N * 96 * 4);  // h0x, then aggs
  float* regionB = (float*)alloc((size_t)3 * N * 96 * 4);  // t/h1, then h2x
  float* pressed = (float*)alloc((size_t)N * 54 * 4);
  float* h1j = (float*)alloc((size_t)N * 34 * 4);
  float* h2j = (float*)alloc((size_t)N * 25 * 4);
  float* h3j = (float*)alloc((size_t)N * 16 * 4);
  float* stats = (float*)alloc(1024 * 4);
  float* hg = (float*)alloc(64 * 75 * 4);
  (void)in_sizes; (void)n_in; (void)out_size; (void)ws_size;

  const int nRowTiles = N / 16;  // 3125

  // 1) f16 copy of node features; pack all WMMA B matrices
  f32_to_f16_kernel<<<gsBlocks((size_t)N * 128), 256, 0, stream>>>(
      node_feats, x16, N * 128);
  pack_conv_kernel<<<gsBlocks(3 * 12 * 6 * 512), 256, 0, stream>>>(conv_w,
                                                                   packConv);
  for (int b = 0; b < 3; ++b)
    pack_rowmajor_kernel<<<gsBlocks(3 * 4 * 512), 256, 0, stream>>>(
        gcn_w + (size_t)b * 96 * 54, packGcn + (size_t)b * 3 * 4 * 512, 96, 54,
        3, 4);
  pack_rowmajor_kernel<<<gsBlocks(2 * 3 * 512), 256, 0, stream>>>(
      jk_w1, packJk1, 54, 34, 2, 3);
  pack_rowmajor_kernel<<<gsBlocks(2 * 2 * 512), 256, 0, stream>>>(
      jk_w2, packJk2, 34, 25, 2, 2);
  pack_rowmajor_kernel<<<gsBlocks(512), 256, 0, stream>>>(jk_w3, packJk3, 25,
                                                          16, 1, 1);

  // 2) dilated convs as WMMA GEMMs -> h0x (regionA) [3][N][96]
  {
    int waves = nRowTiles * 3;
    int blocks = (waves * 32 + 255) / 256;
    conv_wmma_kernel<<<blocks, 256, 0, stream>>>(x16, packConv, conv_b,
                                                 regionA, N);
  }

  // 3) cross-connect t = relu(a)+c -> regionB; BN1 (per-branch) in place
  cross_t_kernel<<<gsBlocks((size_t)3 * N * 96), 256, 0, stream>>>(regionA,
                                                                   regionB, N);
  zero_kernel<<<4, 256, 0, stream>>>(stats, 1024);
  {
    dim3 blk(96, 8);
    dim3 grd((N + 511) / 512);
    for (int b = 0; b < 3; ++b)
      bn_stats_kernel<<<grd, blk, 0, stream>>>(regionB + (size_t)b * N * 96,
                                               stats + b * 96 * 2, N, 96);
  }
  for (int b = 0; b < 3; ++b)
    bn_apply_kernel<<<gsBlocks((size_t)N * 96), 256, 0, stream>>>(
        regionB + (size_t)b * N * 96, regionB + (size_t)b * N * 96,
        stats + b * 96 * 2, bn1_g, bn1_b, N * 96, 96, 1.f / N);

  // 4) GCN: aggregate h1 -> regionA (zeroed), then WMMA gemm -> h2x (regionB)
  zero_kernel<<<gsBlocks((size_t)3 * N * 96), 256, 0, stream>>>(regionA,
                                                                3 * N * 96);
  for (int b = 0; b < 3; ++b)
    edge_agg_kernel<<<gsBlocks((size_t)E * 96), 256, 0, stream>>>(
        edge_src, edge_dst, edge_w, regionB + (size_t)b * N * 96,
        regionA + (size_t)b * N * 96, E, 96);
  {
    int blocks = (nRowTiles * 32 + 255) / 256;
    for (int b = 0; b < 3; ++b)
      wmma_gemm_kernel<4><<<blocks, 256, 0, stream>>>(
          regionA + (size_t)b * N * 96, 96, 96, 3,
          packGcn + (size_t)b * 3 * 4 * 512, 54, gcn_b + b * 54,
          regionB + (size_t)b * N * 54, 54, nRowTiles, 1);
  }

  // 5) press conv over 54-axis + BN2 -> emd (in d_out)
  press_kernel<<<gsBlocks((size_t)N * 54), 256, 0, stream>>>(regionB, press_w,
                                                             press_b, pressed,
                                                             N);
  zero_kernel<<<4, 256, 0, stream>>>(stats, 1024);
  {
    dim3 blk(54, 8);
    dim3 grd((N + 511) / 512);
    bn_stats_kernel<<<grd, blk, 0, stream>>>(pressed, stats, N, 54);
  }
  bn_apply_kernel<<<gsBlocks((size_t)N * 54), 256, 0, stream>>>(
      pressed, emd, stats, bn2_g, bn2_b, N * 54, 54, 1.f / N);

  // 6) JK layers: gconv(x) = relu(agg @ W + b)
  int gemmBlocks = (nRowTiles * 32 + 255) / 256;
  // layer 1: 54 -> 34
  zero_kernel<<<gsBlocks((size_t)N * 54), 256, 0, stream>>>(regionA, N * 54);
  edge_agg_kernel<<<gsBlocks((size_t)E * 54), 256, 0, stream>>>(
      edge_src, edge_dst, edge_w, emd, regionA, E, 54);
  wmma_gemm_kernel<3><<<gemmBlocks, 256, 0, stream>>>(
      regionA, 54, 54, 2, packJk1, 34, jk_b1, h1j, 34, nRowTiles, 1);
  // layer 2: 34 -> 25
  zero_kernel<<<gsBlocks((size_t)N * 34), 256, 0, stream>>>(regionA, N * 34);
  edge_agg_kernel<<<gsBlocks((size_t)E * 34), 256, 0, stream>>>(
      edge_src, edge_dst, edge_w, h1j, regionA, E, 34);
  wmma_gemm_kernel<2><<<gemmBlocks, 256, 0, stream>>>(
      regionA, 34, 34, 2, packJk2, 25, jk_b2, h2j, 25, nRowTiles, 1);
  // layer 3: 25 -> 16
  zero_kernel<<<gsBlocks((size_t)N * 25), 256, 0, stream>>>(regionA, N * 25);
  edge_agg_kernel<<<gsBlocks((size_t)E * 25), 256, 0, stream>>>(
      edge_src, edge_dst, edge_w, h2j, regionA, E, 25);
  wmma_gemm_kernel<1><<<gemmBlocks, 256, 0, stream>>>(
      regionA, 25, 25, 1, packJk3, 16, jk_b3, h3j, 16, nRowTiles, 1);

  // 7) graph readout -> hg [64,75]
  zero_kernel<<<gsBlocks(64 * 75), 256, 0, stream>>>(hg, 64 * 75);
  readout_kernel<<<gsBlocks((size_t)N * 34), 256, 0, stream>>>(graph_ids, h1j,
                                                               hg, N, 34, 0);
  readout_kernel<<<gsBlocks((size_t)N * 25), 256, 0, stream>>>(graph_ids, h2j,
                                                               hg, N, 25, 34);
  readout_kernel<<<gsBlocks((size_t)N * 16), 256, 0, stream>>>(graph_ids, h3j,
                                                               hg, N, 16, 59);

  // 8) BN3 over 64 graphs + classifier head -> logits
  zero_kernel<<<4, 256, 0, stream>>>(stats, 1024);
  {
    dim3 blk(75, 8);
    bn_stats_kernel<<<1, blk, 0, stream>>>(hg, stats, 64, 75);
  }
  classifier_kernel<<<1, 256, 0, stream>>>(hg, stats, bn3_g, bn3_b, cls1_w,
                                           cls1_b, cls2_w, cls2_b, logits);
}
